// Switcher_23570780520756
// MI455X (gfx1250) — compile-verified, matrix operand order checked by reading, not run on hardware
//
#include <hip/hip_runtime.h>
#include <math.h>

#define SEQ      2048
#define BZ       64
#define DIM      1024
#define HID      256
#define DICT_LEN 16

#define MT   64   // seq rows per workgroup
#define KC   32   // layer-1 K chunk (one 16x16x32 wmma k-step)
#define NC2  64   // layer-2 output-column chunk

typedef float          floatx4  __attribute__((ext_vector_type(4)));
typedef float          v8f      __attribute__((ext_vector_type(8)));
typedef __bf16         bf16x2   __attribute__((ext_vector_type(2)));
typedef __bf16         bf16x16  __attribute__((ext_vector_type(16)));
typedef unsigned short ushort8  __attribute__((ext_vector_type(8)));
typedef unsigned int   uintx2   __attribute__((ext_vector_type(2)));

union Frag { bf16x16 bf; ushort8 u[2]; };

// f32 -> bf16: native fptrunc (backend picks v_cvt_*_bf16_f32 when available)
__device__ __forceinline__ unsigned short f2bf(float f) {
  union { __bf16 b; unsigned short u; } c;
  c.b = (__bf16)f;
  return c.u;
}

// pack two f32 into one dword of two bf16
__device__ __forceinline__ unsigned int pack_bf16(float a, float b) {
#if __has_builtin(__builtin_amdgcn_cvt_pk_bf16_f32)
  union { bf16x2 v; unsigned int u; } c;
  c.v = __builtin_amdgcn_cvt_pk_bf16_f32(a, b);
  return c.u;
#else
  union { bf16x2 v; unsigned int u; } c;
  c.v[0] = (__bf16)a;     // fptrunc; vectorizes to v_cvt_pk_bf16_f32 if present
  c.v[1] = (__bf16)b;
  return c.u;
#endif
}

__device__ __forceinline__ void store_bf16x4(unsigned short* d, floatx4 v) {
  uintx2 p;
  p[0] = pack_bf16(v[0], v[1]);
  p[1] = pack_bf16(v[2], v[3]);
  *(uintx2*)d = p;   // 8-byte LDS store
}

__device__ __forceinline__ float gelu_exact(float x) {
  return 0.5f * x * (1.0f + erff(x * 0.70710678118654752440f));
}

__device__ __forceinline__ v8f v8f_zero() {
  v8f z;
  #pragma unroll
  for (int i = 0; i < 8; ++i) z[i] = 0.0f;
  return z;
}

__global__ __launch_bounds__(256)
void switcher_moe_kernel(const float* __restrict__ x,
                         const int*   __restrict__ lang_ids,
                         const float* __restrict__ w1_w,
                         const float* __restrict__ w1_b,
                         const float* __restrict__ w2_w,
                         const float* __restrict__ w2_b,
                         float*       __restrict__ out) {
  // 32 KB persistent H (bf16) + 32 KB staging buffer = 64 KB LDS total
  __shared__ __attribute__((aligned(16))) unsigned short Hs[MT * HID];
  __shared__ __attribute__((aligned(16))) unsigned short Stage[16 * 1024];

  unsigned short* Xs = Stage;            // layer 1: [MT][KC]   = 64*32  (4 KB)
  unsigned short* Ws = Stage + MT * KC;  // layer 1: [HID][KC]  = 256*32 (16 KB)

  const int b    = blockIdx.x;           // batch column
  const int s0   = blockIdx.y * MT;      // seq tile origin
  const int tid  = threadIdx.x;
  const int wave = tid >> 5;
  const int lane = tid & 31;
  const int lhi  = lane >> 4;            // 0 / 1 (half-wave)
  const int llo  = lane & 15;

  // routing: (DICT_LEN-1 - id) <= 3  <=>  id >= DICT_LEN-4
  const int g = (lang_ids[b] >= (DICT_LEN - 4)) ? 1 : 0;
  const float* w1p = w1_w + (size_t)g * HID * DIM;
  const float* b1p = w1_b + (size_t)g * HID;
  const float* w2p = w2_w + (size_t)g * DIM * HID;
  const float* b2p = w2_b + (size_t)g * DIM;

  const float* xcol = x   + (size_t)s0 * (BZ * DIM) + (size_t)b * DIM;
  float*       ocol = out + (size_t)s0 * (BZ * DIM) + (size_t)b * DIM;

  // ---------------- Layer 1: H = gelu(X @ W1^T + b1), kept in LDS ----------------
  const int mt1   = wave & 3;            // M tile 0..3  (rows mt1*16..+16)
  const int nbase = (wave >> 2) * 8;     // first of this wave's 8 N tiles (of 16)

  v8f acc[8];
  #pragma unroll
  for (int j = 0; j < 8; ++j) acc[j] = v8f_zero();

  for (int kc = 0; kc < DIM / KC; ++kc) {
    const int k0 = kc * KC;

    // stage X chunk: 64x32 f32 -> bf16 (512 float4, 2 per thread), streamed (NT)
    #pragma unroll
    for (int i = 0; i < 2; ++i) {
      int f = i * 256 + tid;             // float4 index
      int m = f >> 3, c4 = f & 7;        // 8 float4 per row
      const floatx4* src =
          (const floatx4*)(xcol + (size_t)m * (BZ * DIM) + k0 + c4 * 4);
      floatx4 v = __builtin_nontemporal_load(src);
      if (kc + 1 < DIM / KC) __builtin_prefetch((const float*)src + KC, 0, 1);
      store_bf16x4(Xs + m * KC + c4 * 4, v);
    }
    // stage W1 chunk: 256x32 f32 -> bf16 (2048 float4, 8 per thread), L2-resident
    #pragma unroll
    for (int i = 0; i < 8; ++i) {
      int f = i * 256 + tid;
      int n = f >> 3, c4 = f & 7;
      floatx4 v = *(const floatx4*)(w1p + (size_t)n * DIM + k0 + c4 * 4);
      store_bf16x4(Ws + n * KC + c4 * 4, v);
    }
    __syncthreads();

    // A fragment (this wave's M tile), reused for all 8 N tiles
    Frag fa;
    {
      int off = (mt1 * 16 + llo) * KC + lhi * 8;   // lanes16-31: K 8-15 / 24-31
      fa.u[0] = *(const ushort8*)&Xs[off];
      fa.u[1] = *(const ushort8*)&Xs[off + 16];
    }
    // batch all 8 B-fragment loads (distinct regs), then the 8 WMMAs back-to-back
    Frag fb[8];
    #pragma unroll
    for (int j = 0; j < 8; ++j) {
      int off = ((nbase + j) * 16 + llo) * KC + lhi * 8;
      fb[j].u[0] = *(const ushort8*)&Ws[off];
      fb[j].u[1] = *(const ushort8*)&Ws[off + 16];
    }
    #pragma unroll
    for (int j = 0; j < 8; ++j) {
      acc[j] = __builtin_amdgcn_wmma_f32_16x16x32_bf16(
          false, fa.bf, false, fb[j].bf, (short)0, acc[j], false, false);
    }
    __syncthreads();
  }

  // bias + exact gelu + bf16 -> Hs
  #pragma unroll
  for (int j = 0; j < 8; ++j) {
    int nl = (nbase + j) * 16 + llo;
    float bias = b1p[nl];
    #pragma unroll
    for (int r = 0; r < 8; ++r) {
      int m = mt1 * 16 + lhi * 8 + r;    // C/D layout: m = r (+8 for upper half)
      Hs[m * HID + nl] = f2bf(gelu_exact(acc[j][r] + bias));
    }
  }
  __syncthreads();

  // ---------------- Layer 2: Y = H @ W2^T + b2 ----------------
  unsigned short* W2s = Stage;           // [NC2][HID] = 64*256 (32 KB)
  const int mt2 = wave & 3;
  const int ntb = (wave >> 2) * 2;       // this wave's 2 N tiles (of 4 per chunk)

  // Hs is persistent: hoist this wave's 8 A fragments for the entire layer
  Frag fa2[8];
  #pragma unroll
  for (int ks = 0; ks < HID / 32; ++ks) {
    int off = (mt2 * 16 + llo) * HID + ks * 32 + lhi * 8;
    fa2[ks].u[0] = *(const ushort8*)&Hs[off];
    fa2[ks].u[1] = *(const ushort8*)&Hs[off + 16];
  }

  for (int nc = 0; nc < DIM / NC2; ++nc) {
    const int n0 = nc * NC2;
    __syncthreads();                     // previous chunk's B-fragment reads done

    // stage W2 chunk: 64x256 f32 -> bf16 (4096 float4, 16 per thread)
    #pragma unroll
    for (int i = 0; i < 16; ++i) {
      int f = i * 256 + tid;
      int n = f >> 6, c4 = f & 63;       // 64 float4 per row
      floatx4 v = *(const floatx4*)(w2p + (size_t)(n0 + n) * HID + c4 * 4);
      store_bf16x4(W2s + n * HID + c4 * 4, v);
    }
    __syncthreads();

    #pragma unroll
    for (int j = 0; j < 2; ++j) {
      // batch the 8 B-fragment loads, then the 8 WMMAs
      Frag fb[8];
      #pragma unroll
      for (int ks = 0; ks < HID / 32; ++ks) {
        int off = ((ntb + j) * 16 + llo) * HID + ks * 32 + lhi * 8;
        fb[ks].u[0] = *(const ushort8*)&W2s[off];
        fb[ks].u[1] = *(const ushort8*)&W2s[off + 16];
      }
      v8f a2 = v8f_zero();
      #pragma unroll
      for (int ks = 0; ks < HID / 32; ++ks) {
        a2 = __builtin_amdgcn_wmma_f32_16x16x32_bf16(
            false, fa2[ks].bf, false, fb[ks].bf, (short)0, a2, false, false);
      }
      int o = n0 + (ntb + j) * 16 + llo;
      float bias = b2p[o];
      #pragma unroll
      for (int r = 0; r < 8; ++r) {
        int m = mt2 * 16 + lhi * 8 + r;
        __builtin_nontemporal_store(a2[r] + bias,
                                    ocol + (size_t)m * (BZ * DIM) + o);
      }
    }
  }
}

extern "C" void kernel_launch(void* const* d_in, const int* in_sizes, int n_in,
                              void* d_out, int out_size, void* d_ws, size_t ws_size,
                              hipStream_t stream) {
  (void)in_sizes; (void)n_in; (void)d_ws; (void)ws_size; (void)out_size;
  const float* x        = (const float*)d_in[0];
  const int*   lang_ids = (const int*)  d_in[1];
  const float* w1_w     = (const float*)d_in[2];
  const float* w1_b     = (const float*)d_in[3];
  const float* w2_w     = (const float*)d_in[4];
  const float* w2_b     = (const float*)d_in[5];
  float*       out      = (float*)d_out;

  dim3 grid(BZ, SEQ / MT, 1);   // 64 columns x 32 seq tiles = 2048 WGs
  dim3 block(256, 1, 1);        // 8 wave32
  switcher_moe_kernel<<<grid, block, 0, stream>>>(x, lang_ids, w1_w, w1_b,
                                                  w2_w, w2_b, out);
}
